// RGCN_no_jraph_39290360824134
// MI455X (gfx1250) — matrix operand order repaired and checked by reading, not compiled, not run on hardware
//
#include <hip/hip_runtime.h>
#include <hip/hip_bf16.h>

// ---------- types ----------
typedef __attribute__((ext_vector_type(16))) __bf16 v16bf;
typedef __attribute__((ext_vector_type(8)))  __bf16 v8bf;
typedef __attribute__((ext_vector_type(8)))  float  v8f;

#define N_NODES 20000
#define N_EDGES 640000
#define F_IN    128
#define F_HID   128
#define F_CAT   256
#define OUT_DIM 64
#define N_REL   8

// round-to-nearest-even f32 -> bf16
__device__ __forceinline__ __bf16 f2bf(float f) {
    union { float f; unsigned u; } x; x.f = f;
    unsigned r = x.u + 0x7FFFu + ((x.u >> 16) & 1u);
    unsigned short h = (unsigned short)(r >> 16);
    __bf16 out; __builtin_memcpy(&out, &h, 2);
    return out;
}

// ---------- degree count ----------
__global__ void degree_kernel(const int* __restrict__ snd, const int* __restrict__ rcv,
                              float* __restrict__ deg_s, float* __restrict__ deg_r) {
    int e = blockIdx.x * blockDim.x + threadIdx.x;
    if (e >= N_EDGES) return;
    atomicAdd(deg_s + snd[e], 1.0f);
    atomicAdd(deg_r + rcv[e], 1.0f);
}

// deg -> rsqrt(max(deg,1)) in place, both arrays
__global__ void norm_kernel(float* __restrict__ deg_s, float* __restrict__ deg_r) {
    int n = blockIdx.x * blockDim.x + threadIdx.x;
    if (n >= N_NODES) return;
    deg_s[n] = rsqrtf(fmaxf(deg_s[n], 1.0f));
    deg_r[n] = rsqrtf(fmaxf(deg_r[n], 1.0f));
}

// nodes f32 -> bf16
__global__ void conv_nodes_kernel(const float* __restrict__ src, __bf16* __restrict__ dst, int n) {
    int i = blockIdx.x * blockDim.x + threadIdx.x;
    if (i < n) dst[i] = f2bf(src[i]);
}

// W [R][IN][OUT] f32  ->  Wt [R][OUT][IN] bf16
__global__ void transpose_w_kernel(const float* __restrict__ W, __bf16* __restrict__ Wt,
                                   int IN, int OUT) {
    int gid = blockIdx.x * blockDim.x + threadIdx.x;
    int total = N_REL * IN * OUT;
    if (gid >= total) return;
    int r = gid / (IN * OUT);
    int rem = gid % (IN * OUT);
    int i = rem / OUT;
    int o = rem % OUT;
    Wt[(r * OUT + o) * IN + i] = f2bf(W[gid]);
}

// ---------- WMMA GEMM:  C[20000 x 128] = X[20000 x KDIM] (bf16) @ Wt^T  ----------
// Wt is [128 x KDIM] (row = output column, pre-transposed) for one relation.
// Block: (32,8) = 8 waves; wave ct owns output columns ct*16..ct*16+15 of node tile blockIdx.x.
template <int KDIM>
__global__ void rgcn_gemm_kernel(const __bf16* __restrict__ X,
                                 const __bf16* __restrict__ Wt,
                                 float* __restrict__ C) {
    const int lane = threadIdx.x;       // 0..31 (wave32)
    const int ct   = threadIdx.y;       // 0..7 column tile
    const int nt   = blockIdx.x;        // node tile (16 rows)
    const int m    = lane & 15;
    const int hi   = lane >> 4;         // lane-group select

    const __bf16* arow = X  + (size_t)(nt * 16 + m) * KDIM;
    const __bf16* brow = Wt + (size_t)(ct * 16 + m) * KDIM;

    v8f acc = {};
#pragma unroll
    for (int k0 = 0; k0 < KDIM; k0 += 32) {
        // A fragment (16-bit A 16x32 layout): lanes 0-15: K=k0+0..7 & k0+16..23;
        // lanes 16-31: K=k0+8..15 & k0+24..31
        const __bf16* ap = arow + k0 + hi * 8;
        v8bf a0 = *(const v8bf*)(ap);
        v8bf a1 = *(const v8bf*)(ap + 16);
        // B fragment (16-bit B 32x16 layout): col n = lane&15;
        // lanes 0-15: K=k0+0..7 & k0+16..23; lanes 16-31: K=k0+8..15 & k0+24..31
        const __bf16* bp = brow + k0 + hi * 8;
        v8bf b0 = *(const v8bf*)(bp);
        v8bf b1 = *(const v8bf*)(bp + 16);
        v16bf a, b;
#pragma unroll
        for (int i = 0; i < 8; ++i) {
            a[i] = a0[i]; a[8 + i] = a1[i];
            b[i] = b0[i]; b[8 + i] = b1[i];
        }
        acc = __builtin_amdgcn_wmma_f32_16x16x32_bf16(
            /*neg_a=*/false, a, /*neg_b=*/false, b,
            /*c_mod=*/(short)0, acc, /*reuse_a=*/false, /*reuse_b=*/false);
    }
    // D layout: VGPR v -> row (hi*8 + v), col = lane&15 within tile
    const int col = ct * 16 + (lane & 15);
#pragma unroll
    for (int v = 0; v < 8; ++v) {
        C[(size_t)(nt * 16 + hi * 8 + v) * F_HID + col] = acc[v];
    }
}

// ---------- edge gather / scale / scatter-add for one relation ----------
// 32 threads per edge; lane handles 4 contiguous feature floats.
__global__ void edge_agg_kernel(const int* __restrict__ etypes, const int* __restrict__ snd,
                                const int* __restrict__ rcv,
                                const float* __restrict__ hall,   // [N_NODES x 128] for rel
                                const float* __restrict__ norm_s,
                                float* __restrict__ agg, int rel) {
    int gid  = blockIdx.x * blockDim.x + threadIdx.x;
    int e    = gid >> 5;
    if (e >= N_EDGES) return;
    if (etypes[e] != rel) return;
    int lane = gid & 31;
    int s = snd[e], r = rcv[e];
    float ns = norm_s[s];
    const float4 v = *(const float4*)(hall + (size_t)s * F_HID + lane * 4);
    float* dst = agg + (size_t)r * F_HID + lane * 4;
    atomicAdd(dst + 0, v.x * ns);
    atomicAdd(dst + 1, v.y * ns);
    atomicAdd(dst + 2, v.z * ns);
    atomicAdd(dst + 3, v.w * ns);
}

// ---------- layer-0 finalize: relu(agg*norm_r) -> hcat[:, :128] bf16; nodes -> hcat[:, 128:] ----------
__global__ void finalize_l0_kernel(const float* __restrict__ agg, const float* __restrict__ norm_r,
                                   const float* __restrict__ nodes, __bf16* __restrict__ hcat) {
    int gid = blockIdx.x * blockDim.x + threadIdx.x;
    if (gid >= N_NODES * F_HID) return;
    int n = gid / F_HID, f = gid % F_HID;
    float v = fmaxf(agg[gid] * norm_r[n], 0.0f);
    hcat[(size_t)n * F_CAT + f]          = f2bf(v);
    hcat[(size_t)n * F_CAT + F_HID + f]  = f2bf(nodes[gid]);
}

// ---------- pool: sum over nodes of relu(agg1*norm_r) ----------
// 200 blocks x 128 threads; block b handles 100 nodes; one atomic per (block,feature).
__global__ void pool_kernel(const float* __restrict__ agg, const float* __restrict__ norm_r,
                            float* __restrict__ pooled) {
    int f  = threadIdx.x;
    int n0 = blockIdx.x * 100;
    float s = 0.0f;
#pragma unroll 4
    for (int i = 0; i < 100; ++i) {
        int n = n0 + i;
        s += fmaxf(agg[(size_t)n * F_HID + f] * norm_r[n], 0.0f);
    }
    atomicAdd(pooled + f, s);
}

// ---------- output: out[o] = bout[o] + sum_f (pooled[f]/N) * Wout[f][o] ----------
__global__ void out_kernel(const float* __restrict__ pooled, const float* __restrict__ Wout,
                           const float* __restrict__ bout, float* __restrict__ out) {
    int o = threadIdx.x;
    if (o >= OUT_DIM) return;
    const float inv_n = 1.0f / (float)N_NODES;
    float s = bout[o];
#pragma unroll 8
    for (int f = 0; f < F_HID; ++f)
        s += pooled[f] * inv_n * Wout[f * OUT_DIM + o];
    out[o] = s;
}

// ---------- launch ----------
extern "C" void kernel_launch(void* const* d_in, const int* in_sizes, int n_in,
                              void* d_out, int out_size, void* d_ws, size_t ws_size,
                              hipStream_t stream) {
    const float* nodes    = (const float*)d_in[0];
    const int*   senders  = (const int*)  d_in[1];
    const int*   receivers= (const int*)  d_in[2];
    const int*   etypes   = (const int*)  d_in[3];
    // d_in[4] = n_node (constant 20000, hardcoded)
    const float* kernels0 = (const float*)d_in[5];   // [8,128,128]
    const float* kernels1 = (const float*)d_in[6];   // [8,256,128]
    const float* Wout     = (const float*)d_in[7];   // [128,64]
    const float* bout     = (const float*)d_in[8];   // [64]
    float* out = (float*)d_out;

    // ---- workspace layout (256-byte aligned offsets) ----
    char* ws = (char*)d_ws;
    size_t off = 0;
    auto alloc = [&](size_t bytes) { size_t p = off; off += (bytes + 255) & ~(size_t)255; return p; };
    float*  deg_s   = (float*) (ws + alloc(N_NODES * 4));                 // -> norm_s after norm_kernel
    float*  deg_r   = (float*) (ws + alloc(N_NODES * 4));                 // -> norm_r
    __bf16* nodes_b = (__bf16*)(ws + alloc((size_t)N_NODES * F_IN * 2));
    __bf16* w0t     = (__bf16*)(ws + alloc((size_t)N_REL * F_HID * F_IN * 2));   // [r][out][in]
    __bf16* w1t     = (__bf16*)(ws + alloc((size_t)N_REL * F_HID * F_CAT * 2));  // [r][out][in]
    float*  hallr   = (float*) (ws + alloc((size_t)N_NODES * F_HID * 4)); // per-relation GEMM out
    float*  agg     = (float*) (ws + alloc((size_t)N_NODES * F_HID * 4));
    __bf16* hcat    = (__bf16*)(ws + alloc((size_t)N_NODES * F_CAT * 2));
    float*  pooled  = (float*) (ws + alloc(F_HID * 4));
    (void)ws_size; (void)in_sizes; (void)n_in; (void)out_size;

    // ---- zero accumulators (graph-capture-safe memset nodes) ----
    hipMemsetAsync(deg_s, 0, N_NODES * 4, stream);
    hipMemsetAsync(deg_r, 0, N_NODES * 4, stream);
    hipMemsetAsync(agg,   0, (size_t)N_NODES * F_HID * 4, stream);
    hipMemsetAsync(pooled,0, F_HID * 4, stream);

    // ---- degrees & norms ----
    degree_kernel<<<(N_EDGES + 255) / 256, 256, 0, stream>>>(senders, receivers, deg_s, deg_r);
    norm_kernel<<<(N_NODES + 255) / 256, 256, 0, stream>>>(deg_s, deg_r);

    // ---- precision conversion / weight transpose ----
    conv_nodes_kernel<<<((N_NODES * F_IN) + 255) / 256, 256, 0, stream>>>(nodes, nodes_b, N_NODES * F_IN);
    transpose_w_kernel<<<((N_REL * F_IN * F_HID) + 255) / 256, 256, 0, stream>>>(kernels0, w0t, F_IN,  F_HID);
    transpose_w_kernel<<<((N_REL * F_CAT * F_HID) + 255) / 256, 256, 0, stream>>>(kernels1, w1t, F_CAT, F_HID);

    const int n_tiles    = N_NODES / 16;                 // 1250 (exact)
    const dim3 gemm_blk(32, 8);
    const int edge_blocks = (N_EDGES * 32) / 256;        // 80000

    // ---- layer 0: per-relation GEMM (WMMA bf16) + edge aggregate ----
    for (int r = 0; r < N_REL; ++r) {
        rgcn_gemm_kernel<F_IN><<<n_tiles, gemm_blk, 0, stream>>>(
            nodes_b, w0t + (size_t)r * F_HID * F_IN, hallr);
        edge_agg_kernel<<<edge_blocks, 256, 0, stream>>>(
            etypes, senders, receivers, hallr, deg_s, agg, r);
    }
    finalize_l0_kernel<<<((N_NODES * F_HID) + 255) / 256, 256, 0, stream>>>(agg, deg_r, nodes, hcat);

    // ---- layer 1 ----
    hipMemsetAsync(agg, 0, (size_t)N_NODES * F_HID * 4, stream);
    for (int r = 0; r < N_REL; ++r) {
        rgcn_gemm_kernel<F_CAT><<<n_tiles, gemm_blk, 0, stream>>>(
            hcat, w1t + (size_t)r * F_HID * F_CAT, hallr);
        edge_agg_kernel<<<edge_blocks, 256, 0, stream>>>(
            etypes, senders, receivers, hallr, deg_s, agg, r);
    }

    // ---- pool + output projection ----
    pool_kernel<<<N_NODES / 100, F_HID, 0, stream>>>(agg, deg_r, pooled);
    out_kernel<<<1, 64, 0, stream>>>(pooled, Wout, bout, out);
}
